// biLSTMEncoder_18657337934382
// MI455X (gfx1250) — matrix operand order repaired and checked by reading, not compiled, not run on hardware
//
#include <hip/hip_runtime.h>

// Problem constants (from reference): B=256, SP=SF=128, H=512, V=512
#define B_   256
#define S_   128
#define H_   512
#define G4H_ (4 * H_)

typedef __attribute__((ext_vector_type(16))) __bf16 v16bf;
typedef __attribute__((ext_vector_type(8)))  __bf16 v8bf;
typedef __attribute__((ext_vector_type(8)))  float  v8f;

// ---------------------------------------------------------------------------
// Load helpers matching CDNA5 WMMA VGPR layouts (CDNA5 ISA §7.12.2)
// ---------------------------------------------------------------------------

// B-matrix 32x16 bf16 tile: lane (l%16) holds column n=l%16; lanes<16 hold
// K {0..7,16..23}, lanes>=16 hold K {8..15,24..31} -> two 16B loads at +0, +16.
__device__ inline v16bf load_b16(const __bf16* __restrict__ p) {
  v8bf lo = *(const v8bf*)p;
  v8bf hi = *(const v8bf*)(p + 16);
  return __builtin_shufflevector(lo, hi, 0, 1, 2, 3, 4, 5, 6, 7,
                                 8, 9, 10, 11, 12, 13, 14, 15);
}

// A-matrix 16x32 bf16 tile: lane holds row m=l%16; lanes<16: K {0..7,16..23},
// lanes>=16: K {8..15,24..31} -> two 16B loads at +lhalf*8 and +lhalf*8+16.
__device__ inline v16bf load_a16(const __bf16* __restrict__ row, int kbase,
                                 int lhalf) {
  const __bf16* p = row + kbase + lhalf * 8;
  v8bf lo = *(const v8bf*)p;
  v8bf hi = *(const v8bf*)(p + 16);
  return __builtin_shufflevector(lo, hi, 0, 1, 2, 3, 4, 5, 6, 7,
                                 8, 9, 10, 11, 12, 13, 14, 15);
}

__device__ inline float sigmoidf_(float x) { return 1.0f / (1.0f + __expf(-x)); }

// ---------------------------------------------------------------------------
// Precompute kernels
// ---------------------------------------------------------------------------

__global__ void cvt_f32_to_bf16(const float* __restrict__ src,
                                __bf16* __restrict__ dst, int n) {
  int i = blockIdx.x * blockDim.x + threadIdx.x;
  if (i < n) dst[i] = (__bf16)src[i];
}

// Xbf[dir][t][b][k]: dir0 = past embeddings at time t, dir1 = future
// embeddings at time (S-1-t) (pre-reversed so the step kernel is uniform).
__global__ __launch_bounds__(128) void embed_kernel(
    const int* __restrict__ past, const int* __restrict__ future,
    const float* __restrict__ ne_p, const float* __restrict__ pe_p,
    const float* __restrict__ ne_f, const float* __restrict__ pe_f,
    __bf16* __restrict__ Xbf) {
  int t = blockIdx.x, b = blockIdx.y, dir = blockIdx.z;
  int tt, note;
  const float *ne, *pe;
  if (dir == 0) { tt = t;           note = past[b * S_ + tt];   ne = ne_p; pe = pe_p; }
  else          { tt = S_ - 1 - t;  note = future[b * S_ + tt]; ne = ne_f; pe = pe_f; }
  __bf16* dst = Xbf + (((size_t)dir * S_ + t) * B_ + b) * H_;
  const float* nrow = ne + (size_t)note * H_;
  const float* prow = pe + (size_t)tt * H_;
  for (int k = threadIdx.x; k < H_; k += 128)
    dst[k] = (__bf16)(nrow[k] + prow[k]);
}

// ---------------------------------------------------------------------------
// One LSTM timestep, both directions. grid(16, 32, 2), block(128) = 4 waves.
// Wave w computes the 16x16 tile of gate w for this (batch-tile, h-tile):
//   g = x_t @ Wih^T + h @ Whh^T   (K = 512 + 512, 32 x v_wmma_f32_16x16x32_bf16)
// Then LDS exchange fuses i/f/g/o into the c/h update. h is ping-pong
// buffered across steps (Hin read, Hout written); c is element-owned.
// ---------------------------------------------------------------------------
__global__ __launch_bounds__(128) void lstm_step_kernel(
    const __bf16* __restrict__ Xbf,   // [2][S][B][H]
    const __bf16* __restrict__ Hin,   // [2][B][H]
    __bf16* __restrict__ Hout,        // [2][B][H]
    float* __restrict__ Cws,          // [2][B][H]
    const __bf16* __restrict__ Wih,   // [2][4H][H] bf16
    const __bf16* __restrict__ Whh,   // [2][4H][H] bf16
    const float* __restrict__ bih_f, const float* __restrict__ bhh_f,
    const float* __restrict__ bih_b, const float* __restrict__ bhh_b,
    float* __restrict__ out,          // [2][S][B][H]
    int t) {
  const int mtile = blockIdx.x;   // batch tile 0..15
  const int nh    = blockIdx.y;   // hidden tile 0..31
  const int dir   = blockIdx.z;   // 0..1
  const int tid   = threadIdx.x;
  const int gate  = tid >> 5;     // wave id = gate (i,f,g,o)
  const int lane  = tid & 31;
  const int lhalf = lane >> 4;
  const int lmod  = lane & 15;

  const __bf16* xrow = Xbf + (((size_t)dir * S_ + t) * B_ + mtile * 16 + lmod) * H_;
  const __bf16* hrow = Hin + ((size_t)dir * B_ + mtile * 16 + lmod) * H_;
  const __bf16* wih  = Wih + ((size_t)dir * G4H_ + gate * H_ + nh * 16 + lmod) * H_;
  const __bf16* whh  = Whh + ((size_t)dir * G4H_ + gate * H_ + nh * 16 + lmod) * H_;

  v8f acc = {};
  // Input projection: K = 0..511 over x_t
#pragma unroll
  for (int kt = 0; kt < 16; ++kt) {
    v16bf a = load_a16(xrow, kt * 32, lhalf);
    v16bf b = load_b16(wih + kt * 32 + lhalf * 8);
    acc = __builtin_amdgcn_wmma_f32_16x16x32_bf16(false, a, false, b,
                                                  (short)0, acc, false, false);
  }
  // Recurrent projection: K = 0..511 over h_{t-1}
#pragma unroll
  for (int kt = 0; kt < 16; ++kt) {
    v16bf a = load_a16(hrow, kt * 32, lhalf);
    v16bf b = load_b16(whh + kt * 32 + lhalf * 8);
    acc = __builtin_amdgcn_wmma_f32_16x16x32_bf16(false, a, false, b,
                                                  (short)0, acc, false, false);
  }

  // C/D layout: VGPR r -> M = r + 8*lhalf, N = lmod
  __shared__ float lds[4][16][16];
#pragma unroll
  for (int r = 0; r < 8; ++r) lds[gate][r + lhalf * 8][lmod] = acc[r];
  __syncthreads();

  const float* bih = dir ? bih_b : bih_f;
  const float* bhh = dir ? bhh_b : bhh_f;

#pragma unroll
  for (int e = 0; e < 2; ++e) {
    int idx = tid * 2 + e;            // 0..255 within 16x16 tile
    int m = idx >> 4, n = idx & 15;
    int b  = mtile * 16 + m;
    int hc = nh * 16 + n;
    float gi = lds[0][m][n] + bih[0 * H_ + hc] + bhh[0 * H_ + hc];
    float gf = lds[1][m][n] + bih[1 * H_ + hc] + bhh[1 * H_ + hc];
    float gg = lds[2][m][n] + bih[2 * H_ + hc] + bhh[2 * H_ + hc];
    float go = lds[3][m][n] + bih[3 * H_ + hc] + bhh[3 * H_ + hc];
    float si = sigmoidf_(gi);
    float sf = sigmoidf_(gf);
    float so = sigmoidf_(go);
    float tg = tanhf(gg);
    size_t cidx = ((size_t)dir * B_ + b) * H_ + hc;
    float c = sf * Cws[cidx] + si * tg;
    Cws[cidx] = c;
    float h = so * tanhf(c);
    Hout[cidx] = (__bf16)h;
    out[(((size_t)dir * S_ + t) * B_ + b) * H_ + hc] = h;
  }
}

// ---------------------------------------------------------------------------
// Workspace layout (bytes):
//   Wih_bf : 2*4H*H bf16   =  4 MB
//   Whh_bf : 2*4H*H bf16   =  4 MB
//   Xbf    : 2*S*B*H bf16  = 64 MB
//   Hbf[2] : 2*2*B*H bf16  =  1 MB   (ping-pong)
//   Cws    : 2*B*H f32     =  1 MB
// total ~74 MB
// ---------------------------------------------------------------------------
extern "C" void kernel_launch(void* const* d_in, const int* in_sizes, int n_in,
                              void* d_out, int out_size, void* d_ws,
                              size_t ws_size, hipStream_t stream) {
  (void)in_sizes; (void)n_in; (void)out_size; (void)ws_size;

  const int*   past    = (const int*)d_in[0];
  const int*   future  = (const int*)d_in[1];
  const float* ne_p    = (const float*)d_in[2];
  const float* pe_p    = (const float*)d_in[3];
  const float* ne_f    = (const float*)d_in[4];
  const float* pe_f    = (const float*)d_in[5];
  const float* W_ih_f  = (const float*)d_in[6];
  const float* W_hh_f  = (const float*)d_in[7];
  const float* b_ih_f  = (const float*)d_in[8];
  const float* b_hh_f  = (const float*)d_in[9];
  const float* W_ih_b  = (const float*)d_in[10];
  const float* W_hh_b  = (const float*)d_in[11];
  const float* b_ih_b  = (const float*)d_in[12];
  const float* b_hh_b  = (const float*)d_in[13];
  const float* hs_f    = (const float*)d_in[14];
  const float* cs_f    = (const float*)d_in[15];
  const float* hs_b    = (const float*)d_in[16];
  const float* cs_b    = (const float*)d_in[17];
  float* out = (float*)d_out;

  const size_t WN  = (size_t)G4H_ * H_;  // 4H*H = 1M elements
  const size_t BH  = (size_t)B_ * H_;    // 128K elements
  __bf16* Wih_bf = (__bf16*)d_ws;
  __bf16* Whh_bf = Wih_bf + 2 * WN;
  __bf16* Xbf    = Whh_bf + 2 * WN;
  __bf16* Hbf0   = Xbf + (size_t)2 * S_ * BH / H_ * H_;  // 2*S*B*H
  __bf16* Hbf1   = Hbf0 + 2 * BH;
  float*  Cws    = (float*)(Hbf1 + 2 * BH);

  // Weights and initial hidden state -> bf16
  {
    int n = (int)WN, thr = 256, blk = (n + thr - 1) / thr;
    cvt_f32_to_bf16<<<blk, thr, 0, stream>>>(W_ih_f, Wih_bf, n);
    cvt_f32_to_bf16<<<blk, thr, 0, stream>>>(W_ih_b, Wih_bf + WN, n);
    cvt_f32_to_bf16<<<blk, thr, 0, stream>>>(W_hh_f, Whh_bf, n);
    cvt_f32_to_bf16<<<blk, thr, 0, stream>>>(W_hh_b, Whh_bf + WN, n);
    int m = (int)BH, blk2 = (m + thr - 1) / thr;
    cvt_f32_to_bf16<<<blk2, thr, 0, stream>>>(hs_f, Hbf0, m);
    cvt_f32_to_bf16<<<blk2, thr, 0, stream>>>(hs_b, Hbf0 + BH, m);
  }
  // Initial cell state (fp32 copy)
  hipMemcpyAsync(Cws, cs_f, BH * sizeof(float), hipMemcpyDeviceToDevice, stream);
  hipMemcpyAsync(Cws + BH, cs_b, BH * sizeof(float), hipMemcpyDeviceToDevice,
                 stream);

  // Embedding + positional add -> bf16, backward direction pre-reversed
  embed_kernel<<<dim3(S_, B_, 2), 128, 0, stream>>>(past, future, ne_p, pe_p,
                                                    ne_f, pe_f, Xbf);

  // Time recursion: one kernel per step, ping-pong hidden-state buffers.
  dim3 grid(B_ / 16, H_ / 16, 2);  // (16, 32, 2)
  for (int t = 0; t < S_; ++t) {
    const __bf16* Hin = (t & 1) ? Hbf1 : Hbf0;
    __bf16* Hout      = (t & 1) ? Hbf0 : Hbf1;
    lstm_step_kernel<<<grid, 128, 0, stream>>>(Xbf, Hin, Hout, Cws, Wih_bf,
                                               Whh_bf, b_ih_f, b_hh_f, b_ih_b,
                                               b_hh_b, out, t);
  }
}
